// LSTMModule_79121887527479
// MI455X (gfx1250) — matrix operand order repaired and checked by reading, not compiled, not run on hardware
//
#include <hip/hip_runtime.h>
#include <hip/hip_bf16.h>
#include <stdint.h>

// ---------------------------------------------------------------------------
// 2-layer LSTM on MI455X (gfx1250, wave32, WMMA 16x16x32 bf16, f32 accum)
//
//   x = emb[input]                       (gather, f32 -> bf16 packed)
//   G0 = x @ Wx[0]                       (big WMMA GEMM, hoisted out of scan)
//   per t: gates0 = G0[t] + h0 @ Wh0 + b ; LSTM cell  (M=8 padded to 16)
//          gates1 = [h0_t ; h1] @ [Wx1;Wh1] + b ; LSTM cell
//   out = tops @ Wd + bd                 (big WMMA GEMM, 134 GFLOP, bf16)
//
// All matrices are pre-packed into the exact WMMA per-lane fragment layout so
// every fragment load is two contiguous 16B loads.  Big GEMMs stage tiles to
// LDS with gfx1250 async global->LDS copies (ASYNCcnt), double buffered.
// ---------------------------------------------------------------------------

#define VOCAB  32000
#define EMB    1024
#define HID    1024
#define GCOLS  4096     // 4*HID
#define BATCH  8
#define SEQLEN 256
#define TB     2048     // SEQLEN*BATCH
#define OUTV   32000

typedef __attribute__((ext_vector_type(16))) __bf16 v16bf;
typedef __attribute__((ext_vector_type(8)))  float  v8f;

#if __has_builtin(__builtin_amdgcn_global_load_async_to_lds_b128)
#define HAVE_ASYNC 1
// Builtin signature (from probe): (as1 int4*, as3 int4*, imm offset, imm cpol)
typedef __attribute__((ext_vector_type(4))) int i4vec;
typedef __attribute__((address_space(1))) i4vec gi4;
typedef __attribute__((address_space(3))) i4vec li4;
// generic->as1/as3 via integer round trip (LDS offset lives in the low 32 bits)
#define ASYNC_CP16(gsrc, ldst)                                              \
  __builtin_amdgcn_global_load_async_to_lds_b128(                           \
      (gi4*)(unsigned long long)(gsrc),                                     \
      (li4*)(unsigned)(unsigned long long)(ldst), 0, 0)
#if __has_builtin(__builtin_amdgcn_s_wait_asynccnt)
#define WAIT_ASYNC(n) __builtin_amdgcn_s_wait_asynccnt(n)
#else
#define WAIT_ASYNC(n) asm volatile("s_wait_asynccnt %0" ::"n"(n) : "memory")
#endif
#else
#define HAVE_ASYNC 0
#endif

__device__ __forceinline__ __bf16 f2bf(float x) {
  unsigned u = __builtin_bit_cast(unsigned, x);
  u += 0x7FFFu + ((u >> 16) & 1u);                 // round-to-nearest-even
  unsigned short h = (unsigned short)(u >> 16);
  return __builtin_bit_cast(__bf16, h);
}

__device__ __forceinline__ float sigm(float x) { return 1.0f / (1.0f + __expf(-x)); }

// A-matrix packed layout (16x32 bf16 block -> 512 elems): lane L holds M=L%16,
// element e maps K = e + (e>=8?8:0) + (L>=16?8:0).  Block order: mb*KB + kb.
__device__ __forceinline__ void writeA(__bf16* p, int KBdst, int kbOff,
                                       int m, int col, __bf16 v) {
  int kb = kbOff + (col >> 5);
  int kk = col & 31;
  int hi = (kk >> 3) & 1;
  int e  = (kk & 7) + ((kk & 16) ? 8 : 0);
  int laneM = (m & 15) + (hi << 4);
  p[((size_t)(m >> 4) * KBdst + kb) * 512 + laneM * 16 + e] = v;
}

__device__ __forceinline__ v8f wmma_bf16(v16bf a, v16bf b, v8f c) {
  return __builtin_amdgcn_wmma_f32_16x16x32_bf16(false, a, false, b,
                                                 (short)0, c, false, false);
}

// ---------------------------------------------------------------------------
// Weight pack: src f32 row-major [K,N] -> dst bf16, B-fragment layout:
// 32x16 block (kb,nb) stored contiguously; lane L = K-half*16 + (n-n0),
// element e -> K = kb*32 + (L/16)*8 + (e&7) + (e>=8?16:0).  Block order: nb*KBtot + kbOff + kb.
// ---------------------------------------------------------------------------
__global__ void pack_w_kernel(const float* __restrict__ src, __bf16* __restrict__ dst,
                              int K, int N, int KBtot, int kbOff) {
  const int KBl = K >> 5;
  const size_t total = (size_t)K * N;
  for (size_t p = (size_t)blockIdx.x * blockDim.x + threadIdx.x; p < total;
       p += (size_t)gridDim.x * blockDim.x) {
    size_t blk = p >> 9;
    int r = (int)(p & 511);
    int L = r >> 4, e = r & 15;
    int nb = (int)(blk / KBl), kbl = (int)(blk % KBl);
    int n = nb * 16 + (L & 15);
    int k = kbl * 32 + ((L >> 4) << 3) + (e & 7) + ((e >> 3) << 4);
    dst[((size_t)nb * KBtot + kbOff + kbl) * 512 + r] = f2bf(src[(size_t)k * N + n]);
  }
}

__global__ void bias_sum_kernel(const float* __restrict__ bx, const float* __restrict__ bh,
                                float* __restrict__ dst, int n) {
  int i = blockIdx.x * blockDim.x + threadIdx.x;
  if (i < n) dst[i] = bx[i] + bh[i];
}

__global__ void zero32_kernel(unsigned* __restrict__ p, int n) {
  for (int i = blockIdx.x * blockDim.x + threadIdx.x; i < n; i += gridDim.x * blockDim.x)
    p[i] = 0u;
}

// Embedding gather: row m = t*BATCH + b of xs[T,B,E]; write A-packed bf16.
__global__ void gather_embed_kernel(const int* __restrict__ tok, const float* __restrict__ emb,
                                    __bf16* __restrict__ xp) {
  const int m = blockIdx.x;
  const int t = m >> 3, b = m & 7;
  const int token = tok[b * SEQLEN + t];
  const float* src = emb + (size_t)token * EMB;
#pragma unroll
  for (int j = 0; j < 4; ++j) {
    int k = threadIdx.x + j * 256;
    writeA(xp, EMB >> 5, 0, m, k, f2bf(src[k]));
  }
}

// ---------------------------------------------------------------------------
// Tiled WMMA GEMM: C[M,N](f32) = Ap(bf16,packed) @ Bp(bf16,packed) [+ bias]
// Block tile 128x128, 8 waves (4M x 2N), each wave 2x4 fragments
// -> 8 WMMAs per k-step per wave, ~64 FLOP per L2 byte.
// Tiles staged to LDS via async global->LDS (double buffered) when available.
// REMAP: row m = t*8+b  ->  output row b*SEQLEN + t   ([T,B] -> [B,T]).
// ---------------------------------------------------------------------------
template <bool REMAP, bool BIAS>
__global__ __launch_bounds__(256)
void gemm_bf16_kernel(const __bf16* __restrict__ Ap, const __bf16* __restrict__ Bp,
                      float* __restrict__ C, const float* __restrict__ bias,
                      int KB, int ldC) {
  // per buffer: A = 8 blocks (4096 bf16), B = 8 blocks (4096 bf16)
  __shared__ __align__(32) __bf16 lds[2 * 8192];
  const int tid  = threadIdx.x;
  const int lane = tid & 31;
  const int wave = tid >> 5;
  const int wm = wave >> 1, wn = wave & 1;   // 4 waves in M, 2 in N
  const int mb0 = blockIdx.y * 8;            // 16-row block base
  const int nb0 = blockIdx.x * 8;            // 16-col block base

  v8f acc[2][4] = {};

#if HAVE_ASYNC
  // Each stage: 4 async b128 per thread (2 A + 2 B) == 4 ASYNCcnt ops per wave.
  auto issue = [&](int kb, int buf) {
    __bf16* l = lds + buf * 8192;
#pragma unroll
    for (int c = 0; c < 2; ++c) {
      int f   = tid + c * 256;               // uint4 index 0..511
      int blk = f >> 6;
      int off = (f & 63) << 3;               // element offset in block
      ASYNC_CP16(Ap + ((size_t)(mb0 + blk) * KB + kb) * 512 + off, l + f * 8);
      ASYNC_CP16(Bp + ((size_t)(nb0 + blk) * KB + kb) * 512 + off, l + 4096 + f * 8);
    }
  };
  issue(0, 0);
#endif

  for (int kb = 0; kb < KB; ++kb) {
    const int cur = kb & 1;
    __bf16* l = lds + cur * 8192;
#if HAVE_ASYNC
    if (kb + 1 < KB) {
      issue(kb + 1, cur ^ 1);
      WAIT_ASYNC(4);                          // stage kb landed in LDS
    } else {
      WAIT_ASYNC(0);
    }
    __syncthreads();                          // all waves' stage kb visible
#else
#pragma unroll
    for (int c = 0; c < 2; ++c) {
      int f   = tid + c * 256;
      int blk = f >> 6;
      int off = (f & 63) << 3;
      *(uint4*)(l + f * 8) =
          *(const uint4*)(Ap + ((size_t)(mb0 + blk) * KB + kb) * 512 + off);
      *(uint4*)(l + 4096 + f * 8) =
          *(const uint4*)(Bp + ((size_t)(nb0 + blk) * KB + kb) * 512 + off);
    }
    if (kb + 1 < KB) {
      __builtin_prefetch(Bp + ((size_t)(nb0 + (tid & 7)) * KB + kb + 1) * 512, 0, 1);
      __builtin_prefetch(Ap + ((size_t)(mb0 + (tid & 7)) * KB + kb + 1) * 512, 0, 1);
    }
    __syncthreads();
#endif

    v16bf a0 = *(const v16bf*)(l + ((wm * 2 + 0) * 512 + lane * 16));
    v16bf a1 = *(const v16bf*)(l + ((wm * 2 + 1) * 512 + lane * 16));
    v16bf b0 = *(const v16bf*)(l + 4096 + ((wn * 4 + 0) * 512 + lane * 16));
    v16bf b1 = *(const v16bf*)(l + 4096 + ((wn * 4 + 1) * 512 + lane * 16));
    v16bf b2 = *(const v16bf*)(l + 4096 + ((wn * 4 + 2) * 512 + lane * 16));
    v16bf b3 = *(const v16bf*)(l + 4096 + ((wn * 4 + 3) * 512 + lane * 16));

    acc[0][0] = wmma_bf16(a0, b0, acc[0][0]);
    acc[0][1] = wmma_bf16(a0, b1, acc[0][1]);
    acc[0][2] = wmma_bf16(a0, b2, acc[0][2]);
    acc[0][3] = wmma_bf16(a0, b3, acc[0][3]);
    acc[1][0] = wmma_bf16(a1, b0, acc[1][0]);
    acc[1][1] = wmma_bf16(a1, b1, acc[1][1]);
    acc[1][2] = wmma_bf16(a1, b2, acc[1][2]);
    acc[1][3] = wmma_bf16(a1, b3, acc[1][3]);
    __syncthreads();                          // done reading before re-stage
  }

  // C/D layout: lane L, vgpr v -> row = v + (L>=16 ? 8:0), col = L%16
  const int rowOff    = (lane >> 4) * 8;
  const int colInTile = lane & 15;
#pragma unroll
  for (int i = 0; i < 2; ++i)
#pragma unroll
    for (int j = 0; j < 4; ++j) {
      int mt  = mb0 + wm * 2 + i;
      int nt  = nb0 + wn * 4 + j;
      int col = nt * 16 + colInTile;
      float bv = BIAS ? bias[col] : 0.0f;
#pragma unroll
      for (int v = 0; v < 8; ++v) {
        int row  = mt * 16 + rowOff + v;
        int orow = REMAP ? ((row & 7) * SEQLEN + (row >> 3)) : row;
        C[(size_t)orow * ldC + col] = acc[i][j][v] + bv;
      }
    }
}

// ---------------------------------------------------------------------------
// One LSTM layer step: gates[16,4096] = Ap[16,K] @ Wp[K,4096] (+G0) (+bias),
// then elementwise cell update.  16 WGs x 128 threads; wave g owns gate g
// (i,f,g,o) for 64 hidden columns -> 4 WMMA tiles per wave per k-step.
// Fragments load straight from global (L2-hot) -> compiler pipelines
// global_load_b128 clauses against WMMAs.
// ---------------------------------------------------------------------------
__global__ __launch_bounds__(128)
void lstm_step_kernel(const __bf16* __restrict__ Ap, int KB,
                      const __bf16* __restrict__ Wp,
                      const float* __restrict__ G0row,   // G0 + t*8*4096, or null
                      const float* __restrict__ bias,
                      float* __restrict__ cState,        // [8,HID] f32, in place
                      __bf16* __restrict__ d1, int d1KB, int d1kbOff, int d1mBase,
                      __bf16* __restrict__ d2, int d2KB, int d2kbOff) {
  const int tid  = threadIdx.x;
  const int lane = tid & 31;
  const int g    = tid >> 5;              // wave id == gate id (i,f,g,o)
  const int c0   = blockIdx.x * 64;       // hidden-column slice
  __shared__ float gl[4][8][64];          // gate pre-activations

  v8f acc[4] = {};
  for (int kb = 0; kb < KB; ++kb) {
    v16bf a = *(const v16bf*)(Ap + (size_t)kb * 512 + lane * 16);
#pragma unroll
    for (int nt = 0; nt < 4; ++nt) {
      int nb = g * 64 + (c0 >> 4) + nt;   // gate column tile
      v16bf b = *(const v16bf*)(Wp + ((size_t)nb * KB + kb) * 512 + lane * 16);
      acc[nt] = wmma_bf16(a, b, acc[nt]);
    }
  }

  const bool hasG = (G0row != nullptr);
  if (lane < 16) {                         // lanes 0-15 carry batch rows 0-7
#pragma unroll
    for (int nt = 0; nt < 4; ++nt) {
      int lc   = nt * 16 + lane;
      int gcol = g * HID + c0 + lc;
      float bv = bias[gcol];
#pragma unroll
      for (int v = 0; v < 8; ++v) {
        float x = acc[nt][v] + bv;
        if (hasG) x += G0row[(size_t)v * GCOLS + gcol];
        gl[g][v][lc] = x;
      }
    }
  }
  __syncthreads();

  // elementwise LSTM cell: 8 batches x 64 cols = 512 items over 128 threads
#pragma unroll
  for (int it = 0; it < 4; ++it) {
    int idx = tid + it * 128;
    int b   = idx >> 6;
    int lc  = idx & 63;
    int col = c0 + lc;
    float ig = sigm(gl[0][b][lc]);
    float fg = sigm(gl[1][b][lc]);
    float gg = tanhf(gl[2][b][lc]);
    float og = sigm(gl[3][b][lc]);
    float cP = cState[b * HID + col];
    float cN = fg * cP + ig * gg;
    float h  = og * tanhf(cN);
    cState[b * HID + col] = cN;
    __bf16 hb = f2bf(h);
    writeA(d1, d1KB, d1kbOff, d1mBase + b, col, hb);
    writeA(d2, d2KB, d2kbOff, b, col, hb);
  }
}

// ---------------------------------------------------------------------------
extern "C" void kernel_launch(void* const* d_in, const int* in_sizes, int n_in,
                              void* d_out, int out_size, void* d_ws, size_t ws_size,
                              hipStream_t stream) {
  const int*   tok = (const int*)d_in[0];
  const float* emb = (const float*)d_in[1];
  const float* Wx  = (const float*)d_in[2];
  const float* bx  = (const float*)d_in[3];
  const float* Wh  = (const float*)d_in[4];
  const float* bh  = (const float*)d_in[5];
  const float* Wd  = (const float*)d_in[6];
  const float* bd  = (const float*)d_in[7];
  float* out = (float*)d_out;

  size_t off = 0;
  auto alloc = [&](size_t bytes) -> char* {
    char* p = (char*)d_ws + off;
    off += (bytes + 255) & ~(size_t)255;
    return p;
  };
  __bf16* xp    = (__bf16*)alloc((size_t)TB * EMB * 2);       // packed x (bf16)
  __bf16* Wx0p  = (__bf16*)alloc((size_t)EMB * GCOLS * 2);
  __bf16* Wh0p  = (__bf16*)alloc((size_t)HID * GCOLS * 2);
  __bf16* W1p   = (__bf16*)alloc((size_t)2 * HID * GCOLS * 2); // [Wx1;Wh1], K=2048
  __bf16* Wdp   = (__bf16*)alloc((size_t)HID * OUTV * 2);
  float*  G0    = (float*)alloc((size_t)TB * GCOLS * 4);
  float*  bias2 = (float*)alloc((size_t)2 * GCOLS * 4);
  __bf16* topsP = (__bf16*)alloc((size_t)TB * HID * 2);
  // zero-initialized state region (contiguous)
  char* zbase = (char*)d_ws + off;
  __bf16* h0p[2] = { (__bf16*)alloc(16 * HID * 2),      (__bf16*)alloc(16 * HID * 2) };
  __bf16* A1p[2] = { (__bf16*)alloc(16 * 2 * HID * 2),  (__bf16*)alloc(16 * 2 * HID * 2) };
  float*  c0buf  = (float*)alloc(BATCH * HID * 4);
  float*  c1buf  = (float*)alloc(BATCH * HID * 4);
  size_t zbytes = (char*)d_ws + off - zbase;

  // 1) pack weights to WMMA B-fragment layout (f32 -> bf16)
  pack_w_kernel<<<2048, 256, 0, stream>>>(Wx,                Wx0p, EMB, GCOLS, 32, 0);
  pack_w_kernel<<<2048, 256, 0, stream>>>(Wh,                Wh0p, HID, GCOLS, 32, 0);
  pack_w_kernel<<<2048, 256, 0, stream>>>(Wx + (size_t)EMB * GCOLS, W1p, HID, GCOLS, 64, 0);
  pack_w_kernel<<<2048, 256, 0, stream>>>(Wh + (size_t)HID * GCOLS, W1p, HID, GCOLS, 64, 32);
  pack_w_kernel<<<4096, 256, 0, stream>>>(Wd,                Wdp,  HID, OUTV, 32, 0);
  bias_sum_kernel<<<(2 * GCOLS + 255) / 256, 256, 0, stream>>>(bx, bh, bias2, 2 * GCOLS);
  zero32_kernel<<<256, 256, 0, stream>>>((unsigned*)zbase, (int)(zbytes / 4));

  // 2) embedding gather -> packed bf16 A matrix
  gather_embed_kernel<<<TB, 256, 0, stream>>>(tok, emb, xp);

  // 3) hoisted input projection: G0 = x @ Wx0   [2048,1024]x[1024,4096]
  gemm_bf16_kernel<false, false><<<dim3(GCOLS / 128, TB / 128), 256, 0, stream>>>(
      xp, Wx0p, G0, nullptr, EMB / 32, GCOLS);

  // 4) sequential scan: two tiny WMMA kernels per timestep
  for (int t = 0; t < SEQLEN; ++t) {
    int par = t & 1;
    // layer 0: gates = G0[t] + h0_prev @ Wh0 + b0
    lstm_step_kernel<<<HID / 64, 128, 0, stream>>>(
        h0p[par], HID / 32, Wh0p, G0 + (size_t)t * BATCH * GCOLS, bias2,
        c0buf,
        h0p[par ^ 1], 32, 0, 0,        // h0 for next step
        A1p[par],     64, 0);          // h0 part of layer-1 input (this step)
    // layer 1: gates = [h0_t ; h1_prev] @ [Wx1;Wh1] + b1
    lstm_step_kernel<<<HID / 64, 128, 0, stream>>>(
        A1p[par], 2 * HID / 32, W1p, nullptr, bias2 + GCOLS,
        c1buf,
        topsP,        32, 0, t * BATCH, // top hidden -> projection input
        A1p[par ^ 1], 64, 32);          // h1 part of layer-1 input (next step)
  }

  // 5) output projection: out[B,T,V] = tops @ Wd + bd  (134 GFLOP, bf16 WMMA)
  gemm_bf16_kernel<true, true><<<dim3(OUTV / 128, TB / 128), 256, 0, stream>>>(
      topsP, Wdp, out, bd, HID / 32, OUTV);
}
